// RelativeGlobalAttention_66821101191546
// MI455X (gfx1250) — compile-verified, matrix operand order checked by reading, not run on hardware
//
#include <hip/hip_runtime.h>

// ---------------------------------------------------------------------------
// RelativeGlobalAttention for MI455X (gfx1250), bf16 WMMA path.
//   B=8, S=1024, D=512, H=16, dh=32, qk_coeff = 1/1024, EPS=1e-5
// Pipeline:
//   er_cvt_kernel   : Er f32 -> bf16
//   qkv_proj_kernel : fused QKV GEMM (bf16 WMMA, f32 acc) + q/k normalizations
//   rga_attn_kernel : flash attention with skewed relative term (QEr via WMMA
//                     into LDS, per-element skew gather, online softmax, PV WMMA)
// ---------------------------------------------------------------------------

typedef __attribute__((ext_vector_type(16))) __bf16 v16bf;
typedef __attribute__((ext_vector_type(8)))  __bf16 bf8v;
typedef __attribute__((ext_vector_type(4)))  __bf16 bf4v;
typedef __attribute__((ext_vector_type(8)))  float  v8f;
typedef __attribute__((ext_vector_type(4)))  float  f4v;

#define BB   8
#define SS   1024
#define DD   512
#define HH   16
#define DH   32
#define EPSN 1e-5f

__device__ __forceinline__ v8f wmma_bf16(v16bf a, v16bf b, v8f c) {
  // D = A(16x32 bf16) * B(32x16 bf16) + C(16x16 f32)
  return __builtin_amdgcn_wmma_f32_16x16x32_bf16(false, a, false, b, (short)0, c,
                                                 false, false);
}

// Load one lane's A/B fragment from a row-major bf16 matrix.
// Per ISA: lanes 0-15 hold K chunks {0..7,16..23}; lanes 16-31 {8..15,24..31}.
// Caller passes k0 = (lane>=16 ? 8 : 0) folded into the base offset.
__device__ __forceinline__ v16bf frag_rm(const __bf16* base, int row, int ld, int k0) {
  const __bf16* p = base + (size_t)row * ld + k0;
  bf8v lo = *(const bf8v*)(p);
  bf8v hi = *(const bf8v*)(p + 16);
  v16bf r;
#pragma unroll
  for (int i = 0; i < 8; ++i) { r[i] = lo[i]; r[i + 8] = hi[i]; }
  return r;
}

// A-fragment straight from f32 memory with on-the-fly bf16 conversion.
__device__ __forceinline__ v16bf frag_a_f32(const float* p) {
  f4v a = *(const f4v*)(p);
  f4v b = *(const f4v*)(p + 4);
  f4v c = *(const f4v*)(p + 16);
  f4v d = *(const f4v*)(p + 20);
  v16bf r;
#pragma unroll
  for (int i = 0; i < 4; ++i) {
    r[i]      = (__bf16)a[i];
    r[i + 4]  = (__bf16)b[i];
    r[i + 8]  = (__bf16)c[i];
    r[i + 12] = (__bf16)d[i];
  }
  return r;
}

// ---------------------------------------------------------------------------
__global__ void er_cvt_kernel(const float* __restrict__ Er, __bf16* __restrict__ Er16) {
  int i = blockIdx.x * 256 + threadIdx.x;
  if (i < SS * DH) Er16[i] = (__bf16)Er[i];
}

// ---------------------------------------------------------------------------
// One workgroup = one (batch, head, projection). Computes the full 1024x32
// output strip so the q sequence-axis norm can be finished in-workgroup.
// Outputs: Qn,Kn as (B,H,S,dh) bf16 row-major; Vt as (B,H,dh,S) bf16.
__global__ __launch_bounds__(256)
void qkv_proj_kernel(const float* __restrict__ x,
                     const float* __restrict__ Wq, const float* __restrict__ bq,
                     const float* __restrict__ Wk, const float* __restrict__ bk,
                     const float* __restrict__ Wv, const float* __restrict__ bv,
                     __bf16* __restrict__ Qn, __bf16* __restrict__ Kn,
                     __bf16* __restrict__ Vt) {
  const int h = blockIdx.x, b = blockIdx.y, p = blockIdx.z;
  const int tid = threadIdx.x, lane = tid & 31, wave = tid >> 5;
  const int col = lane & 15, hi = lane >> 4, k0l = hi * 8;

  __shared__ __bf16 wlds[DH * DD];   // 32 KB: one head's weight rows, bf16
  __shared__ float  colsq[DH];       // q column sum-of-squares

  const float* W    = (p == 0) ? Wq : (p == 1) ? Wk : Wv;
  const float* bias = (p == 0) ? bq : (p == 1) ? bk : bv;
  __bf16* dstH = ((p == 0) ? Qn : (p == 1) ? Kn : Vt) + (size_t)(b * HH + h) * SS * DH;

  for (int idx = tid; idx < DH * DD; idx += 256) {
    int r = idx >> 9, c = idx & (DD - 1);
    wlds[idx] = (__bf16)W[(size_t)(h * DH + r) * DD + c];
  }
  if (tid < DH) colsq[tid] = 0.f;
  __syncthreads();

  const float bias0 = bias[h * DH + col];
  const float bias1 = bias[h * DH + 16 + col];
  float csq0 = 0.f, csq1 = 0.f;

  for (int t = 0; t < 8; ++t) {              // 8 row-tiles per wave (64 total)
    const int s0 = (t * 8 + wave) * 16;
    v8f a0 = {}, a1 = {};
    const float* xrow = x + (size_t)(b * SS + s0 + col) * DD;
#pragma unroll
    for (int ks = 0; ks < 16; ++ks) {        // K = 512 in steps of 32
      v16bf af = frag_a_f32(xrow + ks * 32 + k0l);
      v16bf b0 = frag_rm(wlds, col,      DD, ks * 32 + k0l);
      v16bf b1 = frag_rm(wlds, col + 16, DD, ks * 32 + k0l);
      a0 = wmma_bf16(af, b0, a0);
      a1 = wmma_bf16(af, b1, a1);
    }
#pragma unroll
    for (int r = 0; r < 8; ++r) { a0[r] += bias0; a1[r] += bias1; }

    if (p == 2) {                            // V: store transposed (dh,S)
      bf8v p0, p1;
#pragma unroll
      for (int r = 0; r < 8; ++r) { p0[r] = (__bf16)a0[r]; p1[r] = (__bf16)a1[r]; }
      *(bf8v*)(dstH + (size_t)col * SS + s0 + k0l)        = p0;
      *(bf8v*)(dstH + (size_t)(col + 16) * SS + s0 + k0l) = p1;
    } else if (p == 1) {                     // K: row norm over dh
#pragma unroll
      for (int r = 0; r < 8; ++r) {
        float s2 = a0[r] * a0[r] + a1[r] * a1[r];
        s2 += __shfl_xor(s2, 1, 16);
        s2 += __shfl_xor(s2, 2, 16);
        s2 += __shfl_xor(s2, 4, 16);
        s2 += __shfl_xor(s2, 8, 16);
        float sc = 1.f / (sqrtf(s2) + EPSN);
        int s = s0 + hi * 8 + r;
        dstH[(size_t)s * DH + col]      = (__bf16)(a0[r] * sc);
        dstH[(size_t)s * DH + col + 16] = (__bf16)(a1[r] * sc);
      }
    } else {                                 // Q: raw store + column sumsq
#pragma unroll
      for (int r = 0; r < 8; ++r) {
        csq0 += a0[r] * a0[r];
        csq1 += a1[r] * a1[r];
        int s = s0 + hi * 8 + r;
        dstH[(size_t)s * DH + col]      = (__bf16)a0[r];
        dstH[(size_t)s * DH + col + 16] = (__bf16)a1[r];
      }
    }
  }

  if (p == 0) {                              // finish q sequence-axis norm
    atomicAdd(&colsq[col],      csq0);
    atomicAdd(&colsq[col + 16], csq1);
    __threadfence();                         // make raw Qn stores visible
    __syncthreads();
    if (tid < DH) colsq[tid] = 1.f / (sqrtf(colsq[tid]) + EPSN);
    __syncthreads();
    for (int idx = tid * 4; idx < SS * DH; idx += 256 * 4) {
      bf4v v = *(bf4v*)(dstH + idx);
      int c0 = idx & (DH - 1);
#pragma unroll
      for (int e = 0; e < 4; ++e) v[e] = (__bf16)((float)v[e] * colsq[c0 + e]);
      *(bf4v*)(dstH + idx) = v;
    }
  }
}

// ---------------------------------------------------------------------------
// One workgroup = one (b, h, 16-row query block). 4 waves split the key range
// (256 keys each) with online softmax; partials merged through LDS.
__global__ __launch_bounds__(128)
void rga_attn_kernel(const __bf16* __restrict__ Qn, const __bf16* __restrict__ Kn,
                     const __bf16* __restrict__ Vt, const __bf16* __restrict__ Er16,
                     const float* __restrict__ coeff_p, float* __restrict__ out) {
  const int q0 = blockIdx.x * 16, h = blockIdx.y, b = blockIdx.z;
  const int tid = threadIdx.x, lane = tid & 31, wave = tid >> 5;
  const int col = lane & 15, hi = lane >> 4, k0l = hi * 8;
  const float coeff = coeff_p[0];

  __shared__ __bf16 srel[17 * SS];           // QEr rows q0..q0+16 (35 KB)
  __shared__ __bf16 pstage[4][16 * 32];      // per-wave P tile C->A relayout
  __shared__ float  cm[4][16], cl[4][16], cacc[4][16 * 32];

  const __bf16* Qh = Qn + (size_t)(b * HH + h) * SS * DH;
  const __bf16* Kh = Kn + (size_t)(b * HH + h) * SS * DH;
  const __bf16* Vh = Vt + (size_t)(b * HH + h) * DH * SS;

  v16bf qa = frag_rm(Qh, q0 + col, DH, k0l);
  int r1 = q0 + 16 + col; if (r1 > SS - 1) r1 = SS - 1;   // clamp; unused rows
  v16bf qa1 = frag_rm(Qh, r1, DH, k0l);

  // Phase 1: QEr rows q0..q0+16 across all 1024 columns -> LDS (bf16).
  for (int t = wave * 16; t < wave * 16 + 16; ++t) {
    v16bf er = frag_rm(Er16, t * 16 + col, DH, k0l);
    v8f c0 = {}, c1 = {};
    c0 = wmma_bf16(qa,  er, c0);
    c1 = wmma_bf16(qa1, er, c1);
    const int gc = t * 16 + col;
#pragma unroll
    for (int r = 0; r < 8; ++r) srel[(r + 8 * hi) * SS + gc] = (__bf16)c0[r];
    if (hi == 0) srel[16 * SS + gc] = (__bf16)c1[0];      // row q0+16
  }
  __syncthreads();

  // Phase 2: flash loop over this wave's 256 keys in chunks of 32.
  float m[8], lsum[8];
#pragma unroll
  for (int r = 0; r < 8; ++r) { m[r] = -1e30f; lsum[r] = 0.f; }
  v8f o0 = {}, o1 = {};

  const int jbase = wave * 256;
  for (int cch = 0; cch < 8; ++cch) {
    const int jb = jbase + cch * 32;
    __builtin_prefetch(Kh + (size_t)((jb + 32) & (SS - 1)) * DH, 0, 1);

    v16bf kf0 = frag_rm(Kh, jb + col,      DH, k0l);
    v16bf kf1 = frag_rm(Kh, jb + 16 + col, DH, k0l);
    v8f s0 = {}, s1 = {};
    s0 = wmma_bf16(qa, kf0, s0);
    s1 = wmma_bf16(qa, kf1, s1);

    // Add skewed relative term (faithful incl. wrap-around), scale by coeff.
#pragma unroll
    for (int r = 0; r < 8; ++r) {
      const int di = r + 8 * hi;
      const int i  = q0 + di;
      {
        const int j = jb + col, dlt = j - i;
        int ridx = (dlt <= 0) ? di : di + 1;
        int cidx = (dlt <= 0) ? (SS - 1 + dlt) : (dlt - 2);
        cidx = (cidx < 0) ? 0 : cidx;
        float sv = (dlt == 1) ? 0.f : (float)srel[ridx * SS + cidx];
        s0[r] = (s0[r] + sv) * coeff;
      }
      {
        const int j = jb + 16 + col, dlt = j - i;
        int ridx = (dlt <= 0) ? di : di + 1;
        int cidx = (dlt <= 0) ? (SS - 1 + dlt) : (dlt - 2);
        cidx = (cidx < 0) ? 0 : cidx;
        float sv = (dlt == 1) ? 0.f : (float)srel[ridx * SS + cidx];
        s1[r] = (s1[r] + sv) * coeff;
      }
    }

    // Online softmax update (row stats via 16-lane xor-shuffles).
#pragma unroll
    for (int r = 0; r < 8; ++r) {
      float cmax = fmaxf(s0[r], s1[r]);
      cmax = fmaxf(cmax, __shfl_xor(cmax, 1, 16));
      cmax = fmaxf(cmax, __shfl_xor(cmax, 2, 16));
      cmax = fmaxf(cmax, __shfl_xor(cmax, 4, 16));
      cmax = fmaxf(cmax, __shfl_xor(cmax, 8, 16));
      float mn = fmaxf(m[r], cmax);
      float scale = __expf(m[r] - mn);
      m[r] = mn;
      float p0 = __expf(s0[r] - mn);
      float p1 = __expf(s1[r] - mn);
      float rs = p0 + p1;
      rs += __shfl_xor(rs, 1, 16);
      rs += __shfl_xor(rs, 2, 16);
      rs += __shfl_xor(rs, 4, 16);
      rs += __shfl_xor(rs, 8, 16);
      lsum[r] = lsum[r] * scale + rs;
      o0[r] *= scale; o1[r] *= scale;
      s0[r] = p0; s1[r] = p1;
    }

    // P tile C->A relayout via per-wave LDS bounce, then PV WMMAs.
    __bf16* ps = pstage[wave];
#pragma unroll
    for (int r = 0; r < 8; ++r) {
      const int di = r + 8 * hi;
      ps[di * 32 + col]      = (__bf16)s0[r];
      ps[di * 32 + 16 + col] = (__bf16)s1[r];
    }
    asm volatile("s_wait_dscnt 0" ::: "memory");
    v16bf pa  = frag_rm(ps, col, 32, k0l);
    v16bf vf0 = frag_rm(Vh, col,      SS, jb + k0l);
    v16bf vf1 = frag_rm(Vh, col + 16, SS, jb + k0l);
    o0 = wmma_bf16(pa, vf0, o0);
    o1 = wmma_bf16(pa, vf1, o1);
  }

  // Merge 4 per-wave partial softmaxes.
  if (col == 0) {
#pragma unroll
    for (int r = 0; r < 8; ++r) { cm[wave][r + 8 * hi] = m[r]; cl[wave][r + 8 * hi] = lsum[r]; }
  }
#pragma unroll
  for (int r = 0; r < 8; ++r) {
    const int di = r + 8 * hi;
    cacc[wave][di * 32 + col]      = o0[r];
    cacc[wave][di * 32 + 16 + col] = o1[r];
  }
  __syncthreads();

  for (int idx = tid; idx < 512; idx += 128) {
    const int r = idx >> 5, d = idx & 31;
    float M = fmaxf(fmaxf(cm[0][r], cm[1][r]), fmaxf(cm[2][r], cm[3][r]));
    float den = 0.f, val = 0.f;
#pragma unroll
    for (int w = 0; w < 4; ++w) {
      float e = __expf(cm[w][r] - M);
      den += e * cl[w][r];
      val += e * cacc[w][r * 32 + d];
    }
    out[((size_t)b * SS + q0 + r) * DD + h * DH + d] = val / den;
  }
}

// ---------------------------------------------------------------------------
extern "C" void kernel_launch(void* const* d_in, const int* in_sizes, int n_in,
                              void* d_out, int out_size, void* d_ws, size_t ws_size,
                              hipStream_t stream) {
  (void)in_sizes; (void)n_in; (void)out_size; (void)ws_size;
  const float* x     = (const float*)d_in[0];
  const float* Wq    = (const float*)d_in[1];
  const float* bq    = (const float*)d_in[2];
  const float* Wk    = (const float*)d_in[3];
  const float* bk    = (const float*)d_in[4];
  const float* Wv    = (const float*)d_in[5];
  const float* bv    = (const float*)d_in[6];
  const float* Er    = (const float*)d_in[7];
  const float* coeff = (const float*)d_in[8];
  float* out = (float*)d_out;

  char* ws = (char*)d_ws;
  const size_t perT = (size_t)BB * HH * SS * DH * sizeof(__bf16);  // 8 MB each
  __bf16* Qn   = (__bf16*)(ws);
  __bf16* Kn   = (__bf16*)(ws + perT);
  __bf16* Vt   = (__bf16*)(ws + 2 * perT);
  __bf16* Er16 = (__bf16*)(ws + 3 * perT);                         // 64 KB

  er_cvt_kernel<<<dim3((SS * DH + 255) / 256), dim3(256), 0, stream>>>(Er, Er16);
  qkv_proj_kernel<<<dim3(HH, BB, 3), dim3(256), 0, stream>>>(
      x, Wq, bq, Wk, bk, Wv, bv, Qn, Kn, Vt);
  rga_attn_kernel<<<dim3(SS / 16, HH, BB), dim3(128), 0, stream>>>(
      Qn, Kn, Vt, Er16, coeff, out);
}
